// GroupedQueryAttention_8005819039802
// MI455X (gfx1250) — matrix operand order repaired and checked by reading, not compiled, not run on hardware
//
#include <hip/hip_runtime.h>
#include <hip/hip_bf16.h>

typedef __bf16 bf16;
typedef __attribute__((ext_vector_type(16))) __bf16 v16bf;
typedef __attribute__((ext_vector_type(8)))  float  v8f;

#define DEVI __device__ __forceinline__

constexpr int Bc = 2, Sc = 2048, Ec = 2048, HQc = 16, HKVc = 4, Dc = 128, WINc = 512;
constexpr int NQKV  = HQc * Dc + 2 * HKVc * Dc;   // 3072
constexpr int Mrows = Bc * Sc;                    // 4096

// ---------------------------------------------------------------------------
// WMMA fragment loaders (contiguous k = 16*(lane/16) + s for A and B, wave32)
// ---------------------------------------------------------------------------

// A fragment: 16x32 bf16 tile, row-major source with stride `ld`.
// lane<16: row=lane, K = 0..15; lane>=16: row=lane-16, K = 16..31
DEVI v16bf load_frag_a(const bf16* p, int ld) {
  const int lane = threadIdx.x & 31;
  const bf16* q = p + (lane & 15) * ld + ((lane >> 4) << 4);
  v16bf a;
#pragma unroll
  for (int i = 0; i < 8; ++i) a[i] = q[i];
#pragma unroll
  for (int i = 0; i < 8; ++i) a[i + 8] = q[8 + i];
  return a;
}

// B fragment: 32x16 (KxN), sourced from an [N][K] tile (K contiguous per row).
// lane<16: col=lane, K=0..15 ; lane>=16: col=lane-16, K=16..31
DEVI v16bf load_frag_bt(const bf16* p, int ld) {
  const int lane = threadIdx.x & 31;
  const bf16* q = p + (lane & 15) * ld + ((lane >> 4) << 4);
  v16bf b;
#pragma unroll
  for (int i = 0; i < 16; ++i) b[i] = q[i];
  return b;
}

DEVI v8f wmma_bf16(v16bf a, v16bf b, v8f c) {
  return __builtin_amdgcn_wmma_f32_16x16x32_bf16(false, a, false, b, (short)0, c,
                                                 false, false);
}

// ---------------------------------------------------------------------------
// Elementwise prep kernels
// ---------------------------------------------------------------------------
__global__ void cvt_bf16_kernel(const float* __restrict__ src, bf16* __restrict__ dst, int n) {
  int i = blockIdx.x * blockDim.x + threadIdx.x;
  const int stride = gridDim.x * blockDim.x;
  for (; i < n; i += stride) dst[i] = (bf16)src[i];
}

__global__ void pack_bias_kernel(const float* __restrict__ bq, const float* __restrict__ bk,
                                 const float* __restrict__ bv, float* __restrict__ dst) {
  int i = blockIdx.x * blockDim.x + threadIdx.x;
  if (i >= NQKV) return;
  float v;
  if (i < HQc * Dc)            v = bq[i];
  else if (i < (HQc + HKVc) * Dc) v = bk[i - HQc * Dc];
  else                          v = bv[i - (HQc + HKVc) * Dc];
  dst[i] = v;
}

// ---------------------------------------------------------------------------
// GEMM: C[M][N] = A[M][K](bf16) * W[N][K]^T(bf16) + bias[N], f32 out
// Block tile 128x128x32, 256 threads = 8 waves (2 in M x 4 in N),
// each wave computes 64x32 = 4x2 WMMA tiles.
// ---------------------------------------------------------------------------
constexpr int BM = 128, BN = 128, BK = 32;
constexpr int LDT = BK + 8;   // padded LDS stride (20-bank lane stride -> conflict free)

__global__ __launch_bounds__(256)
void gemm_bf16_kernel(const bf16* __restrict__ A, const bf16* __restrict__ W,
                      const float* __restrict__ bias, float* __restrict__ C,
                      int M, int N, int K) {
  __shared__ bf16 sA[BM][LDT];
  __shared__ bf16 sB[BN][LDT];

  const int m0 = blockIdx.y * BM;
  const int n0 = blockIdx.x * BN;
  const int tid = threadIdx.x;
  const int wave = tid >> 5;
  const int wm = (wave & 1) * 64;
  const int wn = (wave >> 1) * 32;
  const int lrow = tid >> 1;
  const int lcol = (tid & 1) * 16;

  v8f acc[4][2] = {};

  for (int k0 = 0; k0 < K; k0 += BK) {
    const bf16* gA = A + (size_t)(m0 + lrow) * K + k0 + lcol;
    const bf16* gB = W + (size_t)(n0 + lrow) * K + k0 + lcol;
    int4 a0 = *(const int4*)(gA);
    int4 a1 = *(const int4*)(gA + 8);
    int4 b0 = *(const int4*)(gB);
    int4 b1 = *(const int4*)(gB + 8);
    if (k0 + BK < K) {           // warm L2/WGP$ for the next k tile
      __builtin_prefetch(gA + BK, 0, 0);
      __builtin_prefetch(gB + BK, 0, 0);
    }
    __syncthreads();             // previous iteration's readers done
    *(int4*)&sA[lrow][lcol]     = a0;
    *(int4*)&sA[lrow][lcol + 8] = a1;
    *(int4*)&sB[lrow][lcol]     = b0;
    *(int4*)&sB[lrow][lcol + 8] = b1;
    __syncthreads();

    v16bf bfrag[2];
#pragma unroll
    for (int j = 0; j < 2; ++j) bfrag[j] = load_frag_bt(&sB[wn + j * 16][0], LDT);
#pragma unroll
    for (int i = 0; i < 4; ++i) {
      v16bf afrag = load_frag_a(&sA[wm + i * 16][0], LDT);
#pragma unroll
      for (int j = 0; j < 2; ++j) acc[i][j] = wmma_bf16(afrag, bfrag[j], acc[i][j]);
    }
  }

  const int lane = tid & 31;
  const int ccol = lane & 15;
  const int crow = (lane >> 4) << 3;
#pragma unroll
  for (int i = 0; i < 4; ++i) {
#pragma unroll
    for (int j = 0; j < 2; ++j) {
      const int col = n0 + wn + j * 16 + ccol;
      const float bvv = bias[col];
      float* dst = C + (size_t)(m0 + wm + i * 16 + crow) * N + col;
#pragma unroll
      for (int r = 0; r < 8; ++r) dst[(size_t)r * N] = acc[i][j][r] + bvv;
    }
  }
}

// ---------------------------------------------------------------------------
// RoPE + head-major repack.  gridDim.y: 0 = Q (scaled by 1/sqrt(D)), 1 = K, 2 = V
// QKV layout: [b*S+s][3072] f32 -> Qr[b][HQ][s][D], Kr/Vb[b][HKV][s][D] bf16
// ---------------------------------------------------------------------------
__global__ void rope_pack_kernel(const float* __restrict__ QKV, bf16* __restrict__ Qr,
                                 bf16* __restrict__ Kr, bf16* __restrict__ Vb) {
  const float kLog = 9.2103403719761836f / 64.f;   // ln(10000)/half
  const int sec = blockIdx.y;
  int idx = blockIdx.x * blockDim.x + threadIdx.x;
  if (sec == 0) {
    if (idx >= Bc * Sc * HQc * (Dc / 2)) return;
    const int t = idx & 63; int rest = idx >> 6;
    const int h = rest % HQc; rest /= HQc;
    const int s = rest % Sc; const int b = rest / Sc;
    const float* src = QKV + (size_t)(b * Sc + s) * NQKV + h * Dc;
    const float x1 = src[t], x2 = src[t + 64];
    const float inv = __expf(-(float)t * kLog);
    float sn, cs; __sincosf((float)s * inv, &sn, &cs);
    const float scale = 0.08838834764831845f;      // 1/sqrt(128), folded into Q
    bf16* dst = Qr + (((size_t)(b * HQc + h)) * Sc + s) * Dc;
    dst[t]      = (bf16)((x1 * cs - x2 * sn) * scale);
    dst[t + 64] = (bf16)((x1 * sn + x2 * cs) * scale);
  } else if (sec == 1) {
    if (idx >= Bc * Sc * HKVc * (Dc / 2)) return;
    const int t = idx & 63; int rest = idx >> 6;
    const int h = rest % HKVc; rest /= HKVc;
    const int s = rest % Sc; const int b = rest / Sc;
    const float* src = QKV + (size_t)(b * Sc + s) * NQKV + HQc * Dc + h * Dc;
    const float x1 = src[t], x2 = src[t + 64];
    const float inv = __expf(-(float)t * kLog);
    float sn, cs; __sincosf((float)s * inv, &sn, &cs);
    bf16* dst = Kr + (((size_t)(b * HKVc + h)) * Sc + s) * Dc;
    dst[t]      = (bf16)(x1 * cs - x2 * sn);
    dst[t + 64] = (bf16)(x1 * sn + x2 * cs);
  } else {
    if (idx >= Bc * Sc * HKVc * Dc) return;
    const int d = idx & 127; int rest = idx >> 7;
    const int h = rest % HKVc; rest /= HKVc;
    const int s = rest % Sc; const int b = rest / Sc;
    Vb[(((size_t)(b * HKVc + h)) * Sc + s) * Dc + d] =
        (bf16)QKV[(size_t)(b * Sc + s) * NQKV + (HQc + HKVc) * Dc + h * Dc + d];
  }
}

// ---------------------------------------------------------------------------
// Flash attention with sliding window (causal, i-j <= 512).
// Block = 64 q rows of one (b, q-head); 4 waves, each owns a 16-row strip.
// Key chunks of 64 staged in LDS; V staged transposed ([d][key]).
// ---------------------------------------------------------------------------
__global__ __launch_bounds__(256)
void attn_kernel(const bf16* __restrict__ Qr, const bf16* __restrict__ Kr,
                 const bf16* __restrict__ Vb, bf16* __restrict__ Ao) {
  __shared__ bf16 sK[64][Dc + 8];    // [key][d]
  __shared__ bf16 sVt[Dc][64 + 8];   // [d][key]
  __shared__ bf16 sP[4][16][72];     // per-wave P strip, [row][key]

  const int q0 = blockIdx.x * 64;
  const int h  = blockIdx.y;
  const int b  = blockIdx.z;
  const int hk = h >> 2;             // HQ/HKV = 4
  const int tid = threadIdx.x;
  const int wave = tid >> 5;
  const int lane = tid & 31;
  const int ccol = lane & 15;
  const int crow = (lane >> 4) << 3;
  const int ldk = Dc + 8;

  // Q fragments for this wave's 16 rows: D=128 -> 4 k-slices of 32
  const bf16* qbase = Qr + (((size_t)(b * HQc + h)) * Sc + q0 + wave * 16) * Dc;
  v16bf aq[4];
#pragma unroll
  for (int kk = 0; kk < 4; ++kk) aq[kk] = load_frag_a(qbase + kk * 32, Dc);

  v8f o[8] = {};
  float mrow[8], lsum[8];
#pragma unroll
  for (int r = 0; r < 8; ++r) { mrow[r] = -1e30f; lsum[r] = 0.f; }

  int jlo = q0 - WINc; if (jlo < 0) jlo = 0;
  const int jend = q0 + 63;
  const bf16* kbase = Kr + ((size_t)(b * HKVc + hk)) * Sc * Dc;
  const bf16* vbase = Vb + ((size_t)(b * HKVc + hk)) * Sc * Dc;

  for (int c0 = jlo & ~63; c0 <= jend; c0 += 64) {
    // ---- cooperative stage of K (row-major) and V (transposed) ----
    {
      const int row = tid >> 2;
      const int seg = (tid & 3) * 32;
      const bf16* gk = kbase + (size_t)(c0 + row) * Dc + seg;
      const bf16* gv = vbase + (size_t)(c0 + row) * Dc + seg;
      int4 k0v = *(const int4*)(gk);
      int4 k1v = *(const int4*)(gk + 8);
      int4 k2v = *(const int4*)(gk + 16);
      int4 k3v = *(const int4*)(gk + 24);
      bf16 vtmp[32];
      *(int4*)&vtmp[0]  = *(const int4*)(gv);
      *(int4*)&vtmp[8]  = *(const int4*)(gv + 8);
      *(int4*)&vtmp[16] = *(const int4*)(gv + 16);
      *(int4*)&vtmp[24] = *(const int4*)(gv + 24);
      __syncthreads();                       // previous chunk's readers done
      *(int4*)&sK[row][seg]      = k0v;
      *(int4*)&sK[row][seg + 8]  = k1v;
      *(int4*)&sK[row][seg + 16] = k2v;
      *(int4*)&sK[row][seg + 24] = k3v;
#pragma unroll
      for (int i = 0; i < 32; ++i) sVt[seg + i][row] = vtmp[i];
      __syncthreads();
    }

    // ---- S = Q K^T : 4 key-tiles x 4 k-slices ----
    v8f sacc[4] = {};
#pragma unroll
    for (int ct = 0; ct < 4; ++ct) {
#pragma unroll
      for (int kk = 0; kk < 4; ++kk) {
        v16bf bk_ = load_frag_bt(&sK[ct * 16][kk * 32], ldk);
        sacc[ct] = wmma_bf16(aq[kk], bk_, sacc[ct]);
      }
    }

    // ---- mask + online softmax (rows live in 16-lane halves) ----
    const int baserow = q0 + wave * 16 + crow;
#pragma unroll
    for (int r = 0; r < 8; ++r) {
      const int i = baserow + r;
      float sv[4];
      float cmax = -1e30f;
#pragma unroll
      for (int ct = 0; ct < 4; ++ct) {
        const int j = c0 + ct * 16 + ccol;
        const bool ok = (j <= i) && (i - j <= WINc);
        const float s = ok ? sacc[ct][r] : -1e30f;
        sv[ct] = s;
        cmax = fmaxf(cmax, s);
      }
#pragma unroll
      for (int off = 8; off >= 1; off >>= 1)
        cmax = fmaxf(cmax, __shfl_xor(cmax, off, 16));
      const float mnew = fmaxf(mrow[r], cmax);
      const float csc = __expf(mrow[r] - mnew);
      mrow[r] = mnew;
      float psum = 0.f;
#pragma unroll
      for (int ct = 0; ct < 4; ++ct) {
        const float p = (sv[ct] > -9.0e29f) ? __expf(sv[ct] - mnew) : 0.f;
        psum += p;
        sP[wave][crow + r][ct * 16 + ccol] = (bf16)p;
      }
#pragma unroll
      for (int off = 8; off >= 1; off >>= 1)
        psum += __shfl_xor(psum, off, 16);
      lsum[r] = lsum[r] * csc + psum;
#pragma unroll
      for (int dt = 0; dt < 8; ++dt) o[dt][r] *= csc;
    }

    // ---- O += P V  (per-wave LDS strip; same-wave LDS is in-order) ----
#pragma unroll
    for (int kk = 0; kk < 2; ++kk) {
      v16bf ap = load_frag_a(&sP[wave][0][kk * 32], 72);
#pragma unroll
      for (int dt = 0; dt < 8; ++dt) {
        v16bf bv_ = load_frag_bt(&sVt[dt * 16][kk * 32], 72);
        o[dt] = wmma_bf16(ap, bv_, o[dt]);
      }
    }
  }

  // ---- normalize + store token-major for the output GEMM ----
#pragma unroll
  for (int r = 0; r < 8; ++r) {
    const int i = q0 + wave * 16 + crow + r;
    const float inv = 1.f / lsum[r];
    bf16* dst = Ao + (size_t)(b * Sc + i) * (HQc * Dc) + h * Dc;
#pragma unroll
    for (int dt = 0; dt < 8; ++dt)
      dst[dt * 16 + ccol] = (bf16)(o[dt][r] * inv);
  }
}

// ---------------------------------------------------------------------------
// Launch
// ---------------------------------------------------------------------------
extern "C" void kernel_launch(void* const* d_in, const int* in_sizes, int n_in,
                              void* d_out, int out_size, void* d_ws, size_t ws_size,
                              hipStream_t stream) {
  (void)in_sizes; (void)n_in; (void)out_size; (void)ws_size;
  const float* x  = (const float*)d_in[0];
  const float* Wq = (const float*)d_in[1];
  const float* bq = (const float*)d_in[2];
  const float* Wk = (const float*)d_in[3];
  const float* bk = (const float*)d_in[4];
  const float* Wv = (const float*)d_in[5];
  const float* bv = (const float*)d_in[6];
  const float* Wo = (const float*)d_in[7];
  const float* bo = (const float*)d_in[8];
  float* out = (float*)d_out;

  char* ws = (char*)d_ws;
  size_t off = 0;
  auto take = [&](size_t bytes) {
    char* p = ws + off;
    off += (bytes + 255) & ~(size_t)255;
    return p;
  };
  bf16*  xb    = (bf16*)take((size_t)Mrows * Ec * 2);
  bf16*  Wqkv  = (bf16*)take((size_t)NQKV * Ec * 2);
  bf16*  Wob   = (bf16*)take((size_t)Ec * (HQc * Dc) * 2);
  float* biasq = (float*)take((size_t)NQKV * 4);
  float* QKV   = (float*)take((size_t)Mrows * NQKV * 4);
  bf16*  Qr    = (bf16*)take((size_t)Bc * HQc * Sc * Dc * 2);
  bf16*  Kr    = (bf16*)take((size_t)Bc * HKVc * Sc * Dc * 2);
  bf16*  Vb2   = (bf16*)take((size_t)Bc * HKVc * Sc * Dc * 2);
  bf16*  Ab    = (bf16*)take((size_t)Mrows * HQc * Dc * 2);

  cvt_bf16_kernel<<<4096, 256, 0, stream>>>(x, xb, Mrows * Ec);
  cvt_bf16_kernel<<<4096, 256, 0, stream>>>(Wq, Wqkv, HQc * Dc * Ec);
  cvt_bf16_kernel<<<2048, 256, 0, stream>>>(Wk, Wqkv + (size_t)HQc * Dc * Ec, HKVc * Dc * Ec);
  cvt_bf16_kernel<<<2048, 256, 0, stream>>>(Wv, Wqkv + (size_t)(HQc + HKVc) * Dc * Ec,
                                            HKVc * Dc * Ec);
  cvt_bf16_kernel<<<4096, 256, 0, stream>>>(Wo, Wob, Ec * HQc * Dc);
  pack_bias_kernel<<<(NQKV + 255) / 256, 256, 0, stream>>>(bq, bk, bv, biasq);

  gemm_bf16_kernel<<<dim3(NQKV / BN, Mrows / BM), 256, 0, stream>>>(
      xb, Wqkv, biasq, QKV, Mrows, NQKV, Ec);

  rope_pack_kernel<<<dim3((Bc * Sc * HQc * (Dc / 2) + 255) / 256, 3), 256, 0, stream>>>(
      QKV, Qr, Kr, Vb2);

  attn_kernel<<<dim3(Sc / 64, HQc, Bc), 256, 0, stream>>>(Qr, Kr, Vb2, Ab);

  gemm_bf16_kernel<<<dim3(Ec / BN, Mrows / BM), 256, 0, stream>>>(
      Ab, Wob, bo, out, Mrows, Ec, Ec);
}
